// MultiHeadSelfAttentiondis_2465311228540
// MI455X (gfx1250) — matrix operand order repaired and checked by reading, not compiled, not run on hardware
//
#include <hip/hip_runtime.h>
#include <hip/hip_bf16.h>

typedef __attribute__((ext_vector_type(16))) _Float16 v16h;
typedef __attribute__((ext_vector_type(8)))  float    v8f;

static constexpr int B_  = 8;
static constexpr int D_  = 8;
static constexpr int BD  = 64;     // B*D
static constexpr int S_  = 512;
static constexpr int L_  = 256;
static constexpr int NO  = 768;    // 3*L
static constexpr int DS  = D_ * S_;  // 4096

// ---------------- WMMA f16 16x16x32 fragment helpers (wave32 layouts) -------

// A: 16x32 (MxK) tile, source row-major [rows][ld] f16.
// lane<16: M=lane, halves e0..7 = K k0+0..7, e8..15 = K k0+16..23
// lane>=16: M=lane-16, offsets +8 / +24.
__device__ __forceinline__ v16h frag_a_f16(const _Float16* __restrict__ base, int ld,
                                           int m0, int k0, int lane) {
  const int row  = m0 + (lane & 15);
  const int koff = (lane >> 4) * 8;
  const _Float16* p = base + (size_t)row * ld + k0 + koff;
  v16h a;
#pragma unroll
  for (int j = 0; j < 8; ++j) { a[j] = p[j]; a[8 + j] = p[16 + j]; }
  return a;
}

// A from f32 source (convert to f16 on load).
__device__ __forceinline__ v16h frag_a_f32(const float* __restrict__ base, int ld,
                                           int m0, int k0, int lane) {
  const int row  = m0 + (lane & 15);
  const int koff = (lane >> 4) * 8;
  const float* p = base + (size_t)row * ld + k0 + koff;
  v16h a;
#pragma unroll
  for (int j = 0; j < 8; ++j) { a[j] = (_Float16)p[j]; a[8 + j] = (_Float16)p[16 + j]; }
  return a;
}

// B: 32x16 (KxN). Source stored K-contiguous per column: src[col][ld] (col = N).
// lane<16: N=lane, e0..7 = K k0+0..7, e8..15 = K k0+16..23 ; lane>=16: N=lane-16, offsets +8 / +24.
__device__ __forceinline__ v16h frag_b_kmaj_f16(const _Float16* __restrict__ base, int ld,
                                                int n0, int k0, int lane) {
  const int col  = n0 + (lane & 15);
  const int koff = (lane >> 4) * 8;
  const _Float16* p = base + (size_t)col * ld + k0 + koff;
  v16h b;
#pragma unroll
  for (int j = 0; j < 8; ++j) { b[j] = p[j]; b[8 + j] = p[16 + j]; }
  return b;
}

__device__ __forceinline__ v16h frag_b_kmaj_f32(const float* __restrict__ base, int ld,
                                                int n0, int k0, int lane) {
  const int col  = n0 + (lane & 15);
  const int koff = (lane >> 4) * 8;
  const float* p = base + (size_t)col * ld + k0 + koff;
  v16h b;
#pragma unroll
  for (int j = 0; j < 8; ++j) { b[j] = (_Float16)p[j]; b[8 + j] = (_Float16)p[16 + j]; }
  return b;
}

// ---------------- Kernel 1: QKV projection --------------------------------
// rows = bd*S + s (32768), cols o in [0,768). q pre-scaled by 1/sqrt(L)=1/16.
// v is stored TRANSPOSED per bd: vT[bd][h][t], so the attn.v GEMM gets
// K-contiguous B fragments.
__global__ __launch_bounds__(256) void qkv_kernel(const float* __restrict__ x,
                                                  const float* __restrict__ w,
                                                  const float* __restrict__ bias,
                                                  _Float16* __restrict__ q_h,
                                                  _Float16* __restrict__ k_h,
                                                  _Float16* __restrict__ v_t) {
  const int lane = threadIdx.x & 31;
  const int wave = threadIdx.x >> 5;
  const int tile = blockIdx.x * 8 + wave;      // 2048 * 48 tiles
  const int TN = NO / 16;                      // 48
  const int tm = tile / TN, tn = tile % TN;
  const int m0 = tm * 16, n0 = tn * 16;

  v8f c = {};
  for (int k0 = 0; k0 < L_; k0 += 32) {
    v16h a = frag_a_f32(x, L_, m0, k0, lane);
    v16h b = frag_b_kmaj_f32(w, L_, n0, k0, lane);   // w[o][l]: K-contig per col
    c = __builtin_amdgcn_wmma_f32_16x16x32_f16(false, a, false, b, (short)0, c, false, false);
  }

  const int col  = n0 + (lane & 15);           // o in [0,768)
  const float bc = bias[col];
  const int seg  = col >> 8;                   // 0=q 1=k 2=v  (uniform per wave)
  const int o    = col & 255;
  const float sc = (seg == 0) ? 0.0625f : 1.0f;
#pragma unroll
  for (int r = 0; r < 8; ++r) {
    const int row = m0 + r + ((lane >> 4) << 3);
    const _Float16 val = (_Float16)((c[r] + bc) * sc);
    if (seg == 2) {
      const int bd = row >> 9, s = row & 511;
      v_t[(size_t)bd * L_ * S_ + (size_t)o * S_ + s] = val;
    } else {
      _Float16* dst = (seg == 0) ? q_h : k_h;
      dst[(size_t)row * L_ + o] = val;
    }
  }
}

// ---------------- Kernel 2: logits = q . k^T ------------------------------
__global__ __launch_bounds__(256) void qk_kernel(const _Float16* __restrict__ q_h,
                                                 const _Float16* __restrict__ k_h,
                                                 float* __restrict__ logits) {
  const int lane = threadIdx.x & 31;
  const int wave = threadIdx.x >> 5;
  const int tile = blockIdx.x * 8 + wave;      // 64 * 32 * 32 tiles
  const int bd = tile >> 10;
  const int t2 = tile & 1023;
  const int tm = t2 >> 5, tn = t2 & 31;
  const int m0 = tm * 16, n0 = tn * 16;
  const _Float16* qb = q_h + (size_t)bd * S_ * L_;
  const _Float16* kb = k_h + (size_t)bd * S_ * L_;

  v8f c = {};
  for (int k0 = 0; k0 < L_; k0 += 32) {
    v16h a = frag_a_f16(qb, L_, m0, k0, lane);
    v16h b = frag_b_kmaj_f16(kb, L_, n0, k0, lane);  // k[t][h]: K-contig per col t
    c = __builtin_amdgcn_wmma_f32_16x16x32_f16(false, a, false, b, (short)0, c, false, false);
  }

  float* outb   = logits + (size_t)bd * S_ * S_;
  const int col = n0 + (lane & 15);
#pragma unroll
  for (int r = 0; r < 8; ++r) {
    const int row = m0 + r + ((lane >> 4) << 3);
    outb[(size_t)row * S_ + col] = c[r];
  }
}

// ---------------- Kernel 3: softmax + post-mask + full-row scatter --------
// One block (256 threads) per row (bd, s). Thread t owns columns 2t, 2t+1.
// Mask zeroes |s - t| <= 1 AFTER softmax (reference semantics).
// Writes the ENTIRE 4096-wide block row: zeros in the 7 off-diagonal 512-col
// segments, probabilities in segment d. No separate memset pass needed.
__global__ __launch_bounds__(256) void softmax_kernel(const float* __restrict__ logits,
                                                      float* __restrict__ block_out,
                                                      _Float16* __restrict__ attn_h) {
  __shared__ float red[256];
  const int r   = blockIdx.x;                  // 0..32767
  const int bd  = r >> 9, s = r & 511;
  const int b   = bd >> 3, d = bd & 7;
  const int tid = threadIdx.x;
  const float* row = logits + (size_t)r * S_;

  const float2 xv = ((const float2*)row)[tid]; // cols 2*tid, 2*tid+1
  red[tid] = fmaxf(xv.x, xv.y);
  __syncthreads();
  for (int off = 128; off > 0; off >>= 1) {
    if (tid < off) red[tid] = fmaxf(red[tid], red[tid + off]);
    __syncthreads();
  }
  const float m = red[0];
  __syncthreads();

  const float e0 = __expf(xv.x - m), e1 = __expf(xv.y - m);
  red[tid] = e0 + e1;
  __syncthreads();
  for (int off = 128; off > 0; off >>= 1) {
    if (tid < off) red[tid] += red[tid + off];
    __syncthreads();
  }
  const float inv = 1.0f / red[0];

  const int t0 = 2 * tid, t1 = 2 * tid + 1;
  float p0 = e0 * inv, p1 = e1 * inv;
  int dd0 = s - t0; if (dd0 < 0) dd0 = -dd0;
  int dd1 = s - t1; if (dd1 < 0) dd1 = -dd1;
  if (dd0 <= 1) p0 = 0.0f;
  if (dd1 <= 1) p1 = 0.0f;

  // f16 copy for the attn.v GEMM (packed pair -> single b32 store)
  union { _Float16 h[2]; unsigned u; } pk;
  pk.h[0] = (_Float16)p0;
  pk.h[1] = (_Float16)p1;
  ((unsigned*)attn_h)[(size_t)r * (S_ / 2) + tid] = pk.u;

  // Full block row: block[b, d*S + s, :]
  float2* brow = (float2*)(block_out + ((size_t)b * DS + (size_t)d * S_ + s) * (size_t)DS);
  const float2 pv = make_float2(p0, p1);
  const float2 zv = make_float2(0.0f, 0.0f);
#pragma unroll
  for (int seg = 0; seg < D_; ++seg) {
    brow[seg * (S_ / 2) + tid] = (seg == d) ? pv : zv;
  }
}

// ---------------- Kernel 4: out = attn . v --------------------------------
__global__ __launch_bounds__(256) void av_kernel(const _Float16* __restrict__ attn_h,
                                                 const _Float16* __restrict__ v_t,
                                                 float* __restrict__ out) {
  const int lane = threadIdx.x & 31;
  const int wave = threadIdx.x >> 5;
  const int tile = blockIdx.x * 8 + wave;      // 64 * 32 * 16 tiles
  const int bd = tile >> 9;
  const int t2 = tile & 511;
  const int tm = t2 >> 4, tn = t2 & 15;
  const int m0 = tm * 16, n0 = tn * 16;
  const _Float16* ab = attn_h + (size_t)bd * S_ * S_;
  const _Float16* vb = v_t    + (size_t)bd * L_ * S_;   // vT[h][t]

  v8f c = {};
  for (int k0 = 0; k0 < S_; k0 += 32) {
    v16h a = frag_a_f16(ab, S_, m0, k0, lane);
    v16h b = frag_b_kmaj_f16(vb, S_, n0, k0, lane);     // vT[h][t]: K-contig per col h
    c = __builtin_amdgcn_wmma_f32_16x16x32_f16(false, a, false, b, (short)0, c, false, false);
  }

  float* ob     = out + (size_t)bd * S_ * L_;
  const int col = n0 + (lane & 15);
#pragma unroll
  for (int r = 0; r < 8; ++r) {
    const int row = m0 + r + ((lane >> 4) << 3);
    ob[(size_t)row * L_ + col] = c[r];
  }
}

// ---------------- Launch ---------------------------------------------------
extern "C" void kernel_launch(void* const* d_in, const int* in_sizes, int n_in,
                              void* d_out, int out_size, void* d_ws, size_t ws_size,
                              hipStream_t stream) {
  (void)in_sizes; (void)n_in; (void)out_size; (void)ws_size;
  const float* x    = (const float*)d_in[0];   // [B,D,S,L] f32
  const float* w    = (const float*)d_in[1];   // [768,256] f32
  const float* bias = (const float*)d_in[2];   // [768] f32

  // Workspace layout (112 MB):
  //   [ 0,16MB)  q_h  f16 [BD,S,L]     \  after qk_kernel these two are dead and
  //   [16,32MB)  k_h  f16 [BD,S,L]     /  the 32MB attn_h (f16 [BD,S,S]) aliases them
  //   [32,48MB)  v_t  f16 [BD,L,S]  (v transposed per bd)
  //   [48,112MB) logits f32 [BD,S,S]
  char* ws = (char*)d_ws;
  _Float16* q_h    = (_Float16*)(ws);
  _Float16* k_h    = (_Float16*)(ws + ((size_t)16 << 20));
  _Float16* v_t    = (_Float16*)(ws + ((size_t)32 << 20));
  float*    logits = (float*)   (ws + ((size_t)48 << 20));
  _Float16* attn_h = (_Float16*)(ws);          // aliases q_h+k_h (safe: stream-ordered)

  float* out   = (float*)d_out;                              // [B, D*S, L]
  float* block = out + (size_t)B_ * DS * L_;                 // [B, D*S, D*S]

  // 1) QKV projection: 2048 x 48 tiles, 8 waves/block
  qkv_kernel<<<(2048 * 48) / 8, 256, 0, stream>>>(x, w, bias, q_h, k_h, v_t);

  // 2) Logits: 64 * 32 * 32 tiles
  qk_kernel<<<(64 * 32 * 32) / 8, 256, 0, stream>>>(q_h, k_h, logits);

  // 3) Softmax + mask + full-row block scatter: one block per (bd, s) row
  softmax_kernel<<<BD * S_, 256, 0, stream>>>(logits, block, attn_h);

  // 4) out = attn . v : 64 * 32 * 16 tiles
  av_kernel<<<(64 * 32 * 16) / 8, 256, 0, stream>>>(attn_h, v_t, out);
}
